// GptOssGroupedExperts_3521873183166
// MI455X (gfx1250) — matrix operand order repaired and checked by reading, not compiled, not run on hardware
//
#include <hip/hip_runtime.h>
#include <hip/hip_bf16.h>

// ---- types ----
typedef _Float16 h2  __attribute__((ext_vector_type(2)));
typedef _Float16 h4  __attribute__((ext_vector_type(4)));
typedef _Float16 h8  __attribute__((ext_vector_type(8)));
typedef _Float16 v16h __attribute__((ext_vector_type(16)));
typedef float    v8f  __attribute__((ext_vector_type(8)));

// ---- problem constants ----
#define E_    64
#define TPE_  128
#define DIM_  1024
#define HID_  1024
#define N1_   2048     // 2*HID
#define KSTEP 32
#define PITCH 40       // f16 per LDS tile row: 32 + 8 pad (80 B, 16B aligned)
#define HPITCH 132     // f32 per LDS row of the H tile (bank pad)
#define ABUF  (TPE_ * PITCH)    // halfs per A buffer
#define BBUF  (128  * PITCH)    // halfs per B buffer

__constant__ constexpr float K_ALPHA = 1.702f;
__constant__ constexpr float K_LIMIT = 7.0f;

// Scheduling fence: keep the store phase / prefetch phase / compute phase
// from being interleaved by the backend scheduler, so next-tile global
// loads stay OUTSTANDING across the barrier + WMMA phase.
#if __has_builtin(__builtin_amdgcn_sched_barrier)
#define SCHED_FENCE() __builtin_amdgcn_sched_barrier(0)
#else
#define SCHED_FENCE() asm volatile("" ::: "memory")
#endif

// Build a v16h A/B fragment from two contiguous 16-byte LDS chunks.
__device__ __forceinline__ v16h ld_frag(const _Float16* p, int off_lo, int off_hi) {
    h8 lo = *(const h8*)(p + off_lo);
    h8 hi = *(const h8*)(p + off_hi);
    return __builtin_shufflevector(lo, hi, 0,1,2,3,4,5,6,7,8,9,10,11,12,13,14,15);
}

// =====================================================================
// Kernel 1: h = x @ W1 + b1 ; act = swiglu(h)  (f16 activations to ws)
// Grid: (N1/128, E). Block: 256 = 8 waves (4 M-waves x 2 N-waves).
// Software pipeline: store tile k -> issue loads k+1 -> barrier -> WMMA k.
// =====================================================================
__global__ __launch_bounds__(256)
void moe_mlp1_swiglu(const float* __restrict__ x,
                     const float* __restrict__ w1,
                     const float* __restrict__ b1,
                     _Float16*    __restrict__ act)
{
    // union region: max( double A+B tiles (40960 B), H tile (67584 B) )
    __shared__ __align__(16) unsigned char smem[TPE_ * HPITCH * 4];
    _Float16* As = (_Float16*)smem;                         // [2][ABUF]
    _Float16* Bs = (_Float16*)(smem + 2 * ABUF * 2);        // [2][BBUF]
    float*    Hs = (float*)smem;                            // [128][HPITCH]

    const int e    = blockIdx.y;
    const int n0   = blockIdx.x * 128;            // N-tile base in [0, 2048)
    const int tid  = threadIdx.x;
    const int lane = tid & 31;
    const int wave = tid >> 5;
    const int mw   = wave & 3;                    // 0..3 : 32-row band
    const int nw   = wave >> 2;                   // 0..1 : 64-col band
    const int lh   = lane >> 4;
    const int l16  = lane & 15;

    const float* xe = x  + (size_t)e * TPE_ * DIM_;
    const float* we = w1 + (size_t)e * DIM_ * N1_;

    v8f c[2][4] = {};

    // register prefetch staging
    float4 pa[4];
    float  pb0[8], pb1[8];

    auto prefetch = [&](int k0) {
        #pragma unroll
        for (int i = 0; i < 4; ++i) {
            int s = tid + i * 256; int row = s >> 3; int col = (s & 7) << 2;
            pa[i] = *(const float4*)(xe + (size_t)row * DIM_ + k0 + col);
        }
        #pragma unroll
        for (int i = 0; i < 8; ++i) {
            int q = tid + i * 256; int n = q & 127; int kp = q >> 7;
            pb0[i] = we[(size_t)(k0 + 2*kp    ) * N1_ + n0 + n];
            pb1[i] = we[(size_t)(k0 + 2*kp + 1) * N1_ + n0 + n];
        }
    };

    prefetch(0);
    int buf = 0;
    const int NK = DIM_ / KSTEP;

    for (int kk = 0; kk < NK; ++kk) {
        // ---- phase 1: convert + store prefetched tile into LDS[buf] ----
        _Float16* Ab = As + buf * ABUF;
        _Float16* Bb = Bs + buf * BBUF;
        #pragma unroll
        for (int i = 0; i < 4; ++i) {
            int s = tid + i * 256; int row = s >> 3; int col = (s & 7) << 2;
            h4 hv = { (_Float16)pa[i].x, (_Float16)pa[i].y,
                      (_Float16)pa[i].z, (_Float16)pa[i].w };
            *(h4*)(Ab + row * PITCH + col) = hv;
        }
        #pragma unroll
        for (int i = 0; i < 8; ++i) {
            int q = tid + i * 256; int n = q & 127; int kp = q >> 7;
            h2 hv = { (_Float16)pb0[i], (_Float16)pb1[i] };
            *(h2*)(Bb + n * PITCH + 2 * kp) = hv;
        }
        SCHED_FENCE();
        // ---- phase 2: issue next tile's global loads (stay in flight) ----
        if (kk + 1 < NK) prefetch((kk + 1) * KSTEP);
        SCHED_FENCE();
        __syncthreads();

        // ---- phase 3: fragments + WMMA on LDS[buf] ----
        v16h a[2], b[4];
        #pragma unroll
        for (int mt = 0; mt < 2; ++mt) {
            int row = mw * 32 + mt * 16 + l16;
            a[mt] = ld_frag(Ab + row * PITCH, lh * 8, 16 + lh * 8);
        }
        #pragma unroll
        for (int nt = 0; nt < 4; ++nt) {
            int n = nw * 64 + nt * 16 + l16;
            b[nt] = ld_frag(Bb + n * PITCH, lh * 8, 16 + lh * 8);
        }
        #pragma unroll
        for (int mt = 0; mt < 2; ++mt)
            #pragma unroll
            for (int nt = 0; nt < 4; ++nt)
                c[mt][nt] = __builtin_amdgcn_wmma_f32_16x16x32_f16(
                    false, a[mt], false, b[nt], (short)0, c[mt][nt], false, false);
        buf ^= 1;
    }

    // all tile-buffer reads done before H tile aliases the same LDS
    __syncthreads();

    // ---- bias add, spill h tile to LDS (resolves even/odd lane split) ----
    #pragma unroll
    for (int nt = 0; nt < 4; ++nt) {
        const int coltile = nw * 64 + nt * 16 + l16;
        const float bias = b1[(size_t)e * N1_ + n0 + coltile];
        #pragma unroll
        for (int mt = 0; mt < 2; ++mt)
            #pragma unroll
            for (int r = 0; r < 8; ++r) {
                int row = mw * 32 + mt * 16 + lh * 8 + r;
                Hs[row * HPITCH + coltile] = c[mt][nt][r] + bias;
            }
    }
    __syncthreads();

    // ---- SwiGLU + f16 store: (g0,l0,g1,l1) per float4 ----
    _Float16* acte = act + (size_t)e * TPE_ * HID_ + (n0 >> 1);
    #pragma unroll
    for (int i = 0; i < 16; ++i) {
        int s   = tid + i * 256;                  // 4096 = 128 rows x 32
        int row = s >> 5;
        int jj  = s & 31;
        const float4 hq = *(const float4*)(Hs + row * HPITCH + 4 * jj);
        float g0 = fminf(hq.x, K_LIMIT);
        float l0 = fminf(fmaxf(hq.y, -K_LIMIT), K_LIMIT);
        float g1 = fminf(hq.z, K_LIMIT);
        float l1 = fminf(fmaxf(hq.w, -K_LIMIT), K_LIMIT);
        float a0 = g0 / (1.0f + __expf(-K_ALPHA * g0)) * (l0 + 1.0f);
        float a1 = g1 / (1.0f + __expf(-K_ALPHA * g1)) * (l1 + 1.0f);
        h2 hv = { (_Float16)a0, (_Float16)a1 };
        *(h2*)(acte + (size_t)row * HID_ + 2 * jj) = hv;
    }
}

// =====================================================================
// Kernel 2: out = act @ W2 + b2   (act f16 in ws, out fp32)
// Grid: (DIM/128, E). Block: 256. Same pipeline structure.
// =====================================================================
__global__ __launch_bounds__(256)
void moe_mlp2(const _Float16* __restrict__ act,
              const float*    __restrict__ w2,
              const float*    __restrict__ b2,
              float*          __restrict__ out)
{
    __shared__ __align__(16) _Float16 As[2 * ABUF];
    __shared__ __align__(16) _Float16 Bs[2 * BBUF];

    const int e    = blockIdx.y;
    const int n0   = blockIdx.x * 128;            // N-tile base in [0, 1024)
    const int tid  = threadIdx.x;
    const int lane = tid & 31;
    const int wave = tid >> 5;
    const int mw   = wave & 3;
    const int nw   = wave >> 2;
    const int lh   = lane >> 4;
    const int l16  = lane & 15;

    const _Float16* ae = act + (size_t)e * TPE_ * HID_;
    const float*    we = w2  + (size_t)e * HID_ * DIM_;

    v8f c[2][4] = {};

    h8    pa[2];           // 2 x 16B f16 copies
    float pb0[8], pb1[8];

    auto prefetch = [&](int k0) {
        #pragma unroll
        for (int i = 0; i < 2; ++i) {
            int s = tid + i * 256;                // 512 slots: 128 rows x 4 groups
            int row = s >> 2; int col = (s & 3) << 3;
            pa[i] = *(const h8*)(ae + (size_t)row * HID_ + k0 + col);
        }
        #pragma unroll
        for (int i = 0; i < 8; ++i) {
            int q = tid + i * 256; int n = q & 127; int kp = q >> 7;
            pb0[i] = we[(size_t)(k0 + 2*kp    ) * DIM_ + n0 + n];
            pb1[i] = we[(size_t)(k0 + 2*kp + 1) * DIM_ + n0 + n];
        }
    };

    prefetch(0);
    int buf = 0;
    const int NK = HID_ / KSTEP;

    for (int kk = 0; kk < NK; ++kk) {
        _Float16* Ab = As + buf * ABUF;
        _Float16* Bb = Bs + buf * BBUF;
        #pragma unroll
        for (int i = 0; i < 2; ++i) {
            int s = tid + i * 256; int row = s >> 2; int col = (s & 3) << 3;
            *(h8*)(Ab + row * PITCH + col) = pa[i];
        }
        #pragma unroll
        for (int i = 0; i < 8; ++i) {
            int q = tid + i * 256; int n = q & 127; int kp = q >> 7;
            h2 hv = { (_Float16)pb0[i], (_Float16)pb1[i] };
            *(h2*)(Bb + n * PITCH + 2 * kp) = hv;
        }
        SCHED_FENCE();
        if (kk + 1 < NK) prefetch((kk + 1) * KSTEP);
        SCHED_FENCE();
        __syncthreads();

        v16h a[2], b[4];
        #pragma unroll
        for (int mt = 0; mt < 2; ++mt) {
            int row = mw * 32 + mt * 16 + l16;
            a[mt] = ld_frag(Ab + row * PITCH, lh * 8, 16 + lh * 8);
        }
        #pragma unroll
        for (int nt = 0; nt < 4; ++nt) {
            int n = nw * 64 + nt * 16 + l16;
            b[nt] = ld_frag(Bb + n * PITCH, lh * 8, 16 + lh * 8);
        }
        #pragma unroll
        for (int mt = 0; mt < 2; ++mt)
            #pragma unroll
            for (int nt = 0; nt < 4; ++nt)
                c[mt][nt] = __builtin_amdgcn_wmma_f32_16x16x32_f16(
                    false, a[mt], false, b[nt], (short)0, c[mt][nt], false, false);
        buf ^= 1;
    }

    // ---- bias + direct fp32 store (16-lane rows = 64B contiguous) ----
    float* oute = out + (size_t)e * TPE_ * DIM_;
    #pragma unroll
    for (int nt = 0; nt < 4; ++nt) {
        const int col  = n0 + nw * 64 + nt * 16 + l16;
        const float bias = b2[(size_t)e * DIM_ + col];
        #pragma unroll
        for (int mt = 0; mt < 2; ++mt)
            #pragma unroll
            for (int r = 0; r < 8; ++r) {
                int row = mw * 32 + mt * 16 + lh * 8 + r;
                oute[(size_t)row * DIM_ + col] = c[mt][nt][r] + bias;
            }
    }
}

// =====================================================================
extern "C" void kernel_launch(void* const* d_in, const int* in_sizes, int n_in,
                              void* d_out, int out_size, void* d_ws, size_t ws_size,
                              hipStream_t stream) {
    const float* x  = (const float*)d_in[0];
    const float* w1 = (const float*)d_in[1];
    const float* b1 = (const float*)d_in[2];
    const float* w2 = (const float*)d_in[3];
    const float* b2 = (const float*)d_in[4];
    float* out = (float*)d_out;
    _Float16* act = (_Float16*)d_ws;   // E*TPE*HID f16 = 16 MB workspace

    dim3 blk(256);
    moe_mlp1_swiglu<<<dim3(N1_ / 128, E_), blk, 0, stream>>>(x, w1, b1, act);
    moe_mlp2      <<<dim3(DIM_ / 128, E_), blk, 0, stream>>>(act, w2, b2, out);
}
